// InfoNCELoss_74921409511687
// MI455X (gfx1250) — matrix operand order repaired and checked by reading, not compiled, not run on hardware
//
#include <hip/hip_runtime.h>
#include <hip/hip_bf16.h>

typedef __attribute__((ext_vector_type(2)))  float    v2f;
typedef __attribute__((ext_vector_type(8)))  float    v8f;
typedef __attribute__((ext_vector_type(4)))  _Float16 v4h;
typedef __attribute__((ext_vector_type(8)))  _Float16 v8h;
typedef __attribute__((ext_vector_type(16))) _Float16 v16h;

#define DIM     256
#define INV_T   10.0f     // 1 / temperature
#define SHIFT   10.0f     // |logit| <= 1/T after normalization -> fixed LSE shift
#define BM      64        // rows per block (GEMM kernels)
#define BN      128       // cols per column-iteration
#define KC      64        // K chunk staged in LDS
#define LDA     68        // fp32 LDS stride (floats): 272B -> 4-bank skew
#define LDH     72        // f16  LDS stride (halves): 144B -> conflict-free

// ---------------------------------------------------------------------------
// Kernel 1: per-row inverse L2 norm.  One wave32 per row.
// ---------------------------------------------------------------------------
__global__ __launch_bounds__(256) void k_invnorm(const float* __restrict__ x,
                                                 float* __restrict__ invn) {
    const int wave = threadIdx.x >> 5;
    const int lane = threadIdx.x & 31;
    const int row  = blockIdx.x * 8 + wave;
    const float4* xr = (const float4*)(x + (size_t)row * DIM);
    float4 a = xr[lane];
    float4 b = xr[lane + 32];
    float s = a.x * a.x + a.y * a.y + a.z * a.z + a.w * a.w
            + b.x * b.x + b.y * b.y + b.z * b.z + b.w * b.w;
#pragma unroll
    for (int off = 1; off < 32; off <<= 1) s += __shfl_xor(s, off, 32);
    if (lane == 0) invn[row] = 1.0f / fmaxf(sqrtf(s), 1e-12f);
}

// ---------------------------------------------------------------------------
// Kernel 1b: fp32 -> (hi, lo) f16 split.  x = hi + lo + O(2^-22 |x|).
// One float4 per thread.
// ---------------------------------------------------------------------------
__global__ __launch_bounds__(256) void k_split(const float* __restrict__ x,
                                               _Float16* __restrict__ hi,
                                               _Float16* __restrict__ lo) {
    const size_t i4 = (size_t)blockIdx.x * 256 + threadIdx.x;
    float4 v = ((const float4*)x)[i4];
    float vv[4] = {v.x, v.y, v.z, v.w};
    v4h h, l;
#pragma unroll
    for (int j = 0; j < 4; ++j) {
        _Float16 hj = (_Float16)vv[j];
        h[j] = hj;
        l[j] = (_Float16)(vv[j] - (float)hj);
    }
    ((v4h*)hi)[i4] = h;
    ((v4h*)lo)[i4] = l;
}

// ---------------------------------------------------------------------------
// Kernel 2a (primary): fused GEMM + shifted LSE using f16x3 split WMMA.
// Block: 256 threads = 8 waves (4 row-quadrants x 2 col-halves).
// Each block owns 64 rows, sweeps all N columns in 128-col tiles.
// ---------------------------------------------------------------------------
__global__ __launch_bounds__(256) void k_gemm_lse_f16(
        const _Float16* __restrict__ a_hi, const _Float16* __restrict__ a_lo,
        const _Float16* __restrict__ b_hi, const _Float16* __restrict__ b_lo,
        const float* __restrict__ inv1, const float* __restrict__ inv2,
        float* __restrict__ lse, int n) {
    __shared__ _Float16 AsH[BM * LDH];   //  9216 B
    __shared__ _Float16 AsL[BM * LDH];   //  9216 B
    __shared__ _Float16 BsH[BN * LDH];   // 18432 B
    __shared__ _Float16 BsL[BN * LDH];   // 18432 B
    __shared__ float    red[2 * BM];     //   512 B

    const int tid    = threadIdx.x;
    const int wave   = tid >> 5;
    const int lane   = tid & 31;
    const int lane15 = lane & 15;
    const int half   = lane >> 4;
    const int wr     = wave & 3;         // row quadrant (16 rows)
    const int wc     = wave >> 2;        // column half (64 cols)
    const int gRow0  = blockIdx.x * BM;

    // fragment base offsets (halves) within the LDS tiles
    const int aOff = (16 * wr + lane15) * LDH;
    int bOff[4];
#pragma unroll
    for (int t = 0; t < 4; ++t) bOff[t] = (64 * wc + 16 * t + lane15) * LDH;

    float invA[8];
    {
        const int rbase = gRow0 + 16 * wr + 8 * half;
#pragma unroll
        for (int r = 0; r < 8; ++r) invA[r] = inv1[rbase + r] * INV_T;
    }
    float s[8];
#pragma unroll
    for (int r = 0; r < 8; ++r) s[r] = 0.0f;

    const v8f zacc = {0.f, 0.f, 0.f, 0.f, 0.f, 0.f, 0.f, 0.f};

    for (int cb = 0; cb < n; cb += BN) {
        v8f acc[4];
#pragma unroll
        for (int t = 0; t < 4; ++t) acc[t] = zacc;

#pragma unroll
        for (int kc = 0; kc < DIM; kc += KC) {
            __syncthreads();
            // stage A chunk: 64 rows x 64 k, hi+lo  (each: 512 v8h, 2/thread)
#pragma unroll
            for (int i = 0; i < 2; ++i) {
                const int f   = tid + 256 * i;
                const int row = f >> 3, k8 = f & 7;
                const size_t g = (size_t)(gRow0 + row) * DIM + kc + k8 * 8;
                *(v8h*)(AsH + row * LDH + k8 * 8) = *(const v8h*)(a_hi + g);
                *(v8h*)(AsL + row * LDH + k8 * 8) = *(const v8h*)(a_lo + g);
            }
            // stage B chunk: 128 cols x 64 k, hi+lo (each: 1024 v8h, 4/thread)
#pragma unroll
            for (int i = 0; i < 4; ++i) {
                const int f   = tid + 256 * i;
                const int col = f >> 3, k8 = f & 7;
                const size_t g = (size_t)(cb + col) * DIM + kc + k8 * 8;
                *(v8h*)(BsH + col * LDH + k8 * 8) = *(const v8h*)(b_hi + g);
                *(v8h*)(BsL + col * LDH + k8 * 8) = *(const v8h*)(b_lo + g);
            }
            __syncthreads();

            // two K=32 sub-chunks
#pragma unroll
            for (int kb = 0; kb < KC; kb += 32) {
                // A frags: halves hold K 0-15 / 16-31 contiguously
                v8h ah0 = *(const v8h*)(AsH + aOff + kb + 16 * half);
                v8h ah1 = *(const v8h*)(AsH + aOff + kb + 16 * half + 8);
                v8h al0 = *(const v8h*)(AsL + aOff + kb + 16 * half);
                v8h al1 = *(const v8h*)(AsL + aOff + kb + 16 * half + 8);
                v16h aH = __builtin_shufflevector(ah0, ah1, 0,1,2,3,4,5,6,7,8,9,10,11,12,13,14,15);
                v16h aL = __builtin_shufflevector(al0, al1, 0,1,2,3,4,5,6,7,8,9,10,11,12,13,14,15);
#pragma unroll
                for (int t = 0; t < 4; ++t) {
                    // B frags: halves hold K 0-15 / 16-31 contiguously
                    v8h bh0 = *(const v8h*)(BsH + bOff[t] + kb + 16 * half);
                    v8h bh1 = *(const v8h*)(BsH + bOff[t] + kb + 16 * half + 8);
                    v8h bl0 = *(const v8h*)(BsL + bOff[t] + kb + 16 * half);
                    v8h bl1 = *(const v8h*)(BsL + bOff[t] + kb + 16 * half + 8);
                    v16h bH = __builtin_shufflevector(bh0, bh1, 0,1,2,3,4,5,6,7,8,9,10,11,12,13,14,15);
                    v16h bL = __builtin_shufflevector(bl0, bl1, 0,1,2,3,4,5,6,7,8,9,10,11,12,13,14,15);
                    acc[t] = __builtin_amdgcn_wmma_f32_16x16x32_f16(
                        false, aH, false, bH, (short)0, acc[t], false, false);
                    acc[t] = __builtin_amdgcn_wmma_f32_16x16x32_f16(
                        false, aH, false, bL, (short)0, acc[t], false, false);
                    acc[t] = __builtin_amdgcn_wmma_f32_16x16x32_f16(
                        false, aL, false, bH, (short)0, acc[t], false, false);
                }
            }
        }

        // logit = dot * inv1[row] * inv2[col] / T ; s += exp(logit - SHIFT)
#pragma unroll
        for (int t = 0; t < 4; ++t) {
            const float c2 = inv2[cb + 64 * wc + 16 * t + lane15];
#pragma unroll
            for (int r = 0; r < 8; ++r) {
                const float x = fmaf(acc[t][r] * invA[r], c2, -SHIFT);
                s[r] += __expf(x);
            }
        }
    }

#pragma unroll
    for (int r = 0; r < 8; ++r) {
#pragma unroll
        for (int off = 1; off < 16; off <<= 1) s[r] += __shfl_xor(s[r], off, 32);
    }
    if (lane15 == 0) {
        const int lrow = 16 * wr + 8 * half;
#pragma unroll
        for (int r = 0; r < 8; ++r) red[wc * BM + lrow + r] = s[r];
    }
    __syncthreads();
    if (tid < BM) {
        const float tot = red[tid] + red[BM + tid];
        lse[gRow0 + tid] = SHIFT + __logf(tot);
    }
}

// ---------------------------------------------------------------------------
// Kernel 2b (fallback, small ws): fused GEMM + LSE with fp32 WMMA 16x16x4.
// ---------------------------------------------------------------------------
__global__ __launch_bounds__(256) void k_gemm_lse_f32(
        const float* __restrict__ v1, const float* __restrict__ v2,
        const float* __restrict__ inv1, const float* __restrict__ inv2,
        float* __restrict__ lse, int n) {
    __shared__ float As[BM * LDA];
    __shared__ float Bs[BN * LDA];
    __shared__ float red[2 * BM];

    const int tid    = threadIdx.x;
    const int wave   = tid >> 5;
    const int lane   = tid & 31;
    const int lane15 = lane & 15;
    const int half   = lane >> 4;
    const int wr     = wave & 3;
    const int wc     = wave >> 2;
    const int gRow0  = blockIdx.x * BM;

    const int aBase = (16 * wr + lane15) * LDA + 2 * half;
    int bBase[4];
#pragma unroll
    for (int t = 0; t < 4; ++t) bBase[t] = (64 * wc + 16 * t + lane15) * LDA + 2 * half;

    float invA[8];
    {
        const int rbase = gRow0 + 16 * wr + 8 * half;
#pragma unroll
        for (int r = 0; r < 8; ++r) invA[r] = inv1[rbase + r] * INV_T;
    }
    float s[8];
#pragma unroll
    for (int r = 0; r < 8; ++r) s[r] = 0.0f;

    const v8f zacc = {0.f, 0.f, 0.f, 0.f, 0.f, 0.f, 0.f, 0.f};

    for (int cb = 0; cb < n; cb += BN) {
        v8f acc[4];
#pragma unroll
        for (int t = 0; t < 4; ++t) acc[t] = zacc;

#pragma unroll
        for (int kc = 0; kc < DIM; kc += KC) {
            __syncthreads();
#pragma unroll
            for (int i = 0; i < 4; ++i) {
                const int f = tid + 256 * i;
                const int row = f >> 4, kk = f & 15;
                *(float4*)(As + row * LDA + kk * 4) =
                    *(const float4*)(v1 + (size_t)(gRow0 + row) * DIM + kc + kk * 4);
            }
#pragma unroll
            for (int i = 0; i < 8; ++i) {
                const int f = tid + 256 * i;
                const int col = f >> 4, kk = f & 15;
                *(float4*)(Bs + col * LDA + kk * 4) =
                    *(const float4*)(v2 + (size_t)(cb + col) * DIM + kc + kk * 4);
            }
            __syncthreads();

#pragma unroll
            for (int ks = 0; ks < KC / 4; ++ks) {
                v2f a = *(const v2f*)(As + aBase + 4 * ks);
#pragma unroll
                for (int t = 0; t < 4; ++t) {
                    v2f b = *(const v2f*)(Bs + bBase[t] + 4 * ks);
                    acc[t] = __builtin_amdgcn_wmma_f32_16x16x4_f32(
                        false, a, false, b, (short)0, acc[t], false, false);
                }
            }
        }

#pragma unroll
        for (int t = 0; t < 4; ++t) {
            const float c2 = inv2[cb + 64 * wc + 16 * t + lane15];
#pragma unroll
            for (int r = 0; r < 8; ++r) {
                const float x = fmaf(acc[t][r] * invA[r], c2, -SHIFT);
                s[r] += __expf(x);
            }
        }
    }

#pragma unroll
    for (int r = 0; r < 8; ++r) {
#pragma unroll
        for (int off = 1; off < 16; off <<= 1) s[r] += __shfl_xor(s[r], off, 32);
    }
    if (lane15 == 0) {
        const int lrow = 16 * wr + 8 * half;
#pragma unroll
        for (int r = 0; r < 8; ++r) red[wc * BM + lrow + r] = s[r];
    }
    __syncthreads();
    if (tid < BM) {
        const float tot = red[tid] + red[BM + tid];
        lse[gRow0 + tid] = SHIFT + __logf(tot);
    }
}

// ---------------------------------------------------------------------------
// Kernel 3: diagonal logit + per-block partial loss sums (8 rows / block).
// ---------------------------------------------------------------------------
__global__ __launch_bounds__(256) void k_diag_loss(const float* __restrict__ v1,
                                                   const float* __restrict__ v2,
                                                   const float* __restrict__ inv1,
                                                   const float* __restrict__ inv2,
                                                   const float* __restrict__ lse,
                                                   float* __restrict__ partial) {
    __shared__ float wl[8];
    const int wave = threadIdx.x >> 5;
    const int lane = threadIdx.x & 31;
    const int row  = blockIdx.x * 8 + wave;
    const float4* a = (const float4*)(v1 + (size_t)row * DIM);
    const float4* b = (const float4*)(v2 + (size_t)row * DIM);
    float4 a0 = a[lane], a1 = a[lane + 32];
    float4 b0 = b[lane], b1 = b[lane + 32];
    float d = a0.x * b0.x + a0.y * b0.y + a0.z * b0.z + a0.w * b0.w
            + a1.x * b1.x + a1.y * b1.y + a1.z * b1.z + a1.w * b1.w;
#pragma unroll
    for (int off = 1; off < 32; off <<= 1) d += __shfl_xor(d, off, 32);
    if (lane == 0) {
        const float pos = d * inv1[row] * inv2[row] * INV_T;
        wl[wave] = lse[row] - pos;
    }
    __syncthreads();
    if (threadIdx.x == 0) {
        float t = 0.0f;
#pragma unroll
        for (int w = 0; w < 8; ++w) t += wl[w];
        partial[blockIdx.x] = t;
    }
}

// ---------------------------------------------------------------------------
// Kernel 4: deterministic final reduction -> mean loss.
// ---------------------------------------------------------------------------
__global__ __launch_bounds__(256) void k_final(const float* __restrict__ partial,
                                               float* __restrict__ out,
                                               int nPart, int n) {
    __shared__ float sm[256];
    const int t = threadIdx.x;
    float v = 0.0f;
    for (int i = t; i < nPart; i += 256) v += partial[i];
    sm[t] = v;
    __syncthreads();
#pragma unroll
    for (int st = 128; st > 0; st >>= 1) {
        if (t < st) sm[t] += sm[t + st];
        __syncthreads();
    }
    if (t == 0) out[0] = sm[0] / (float)n;
}

// ---------------------------------------------------------------------------
extern "C" void kernel_launch(void* const* d_in, const int* in_sizes, int n_in,
                              void* d_out, int out_size, void* d_ws, size_t ws_size,
                              hipStream_t stream) {
    const float* v1 = (const float*)d_in[0];
    const float* v2 = (const float*)d_in[1];
    const int n = in_sizes[0] / DIM;          // 16384

    float* w       = (float*)d_ws;
    float* inv1    = w;                        // n floats
    float* inv2    = w + n;                    // n floats
    float* lse     = w + 2 * n;                // n floats
    float* partial = w + 3 * n;                // n/8 floats

    const size_t smallFloats = (size_t)3 * n + n / 8;
    const size_t halfElems   = (size_t)n * DIM;          // per f16 array
    const size_t needF16     = smallFloats * sizeof(float) + 4 * halfElems * sizeof(_Float16);

    k_invnorm<<<n / 8, 256, 0, stream>>>(v1, inv1);
    k_invnorm<<<n / 8, 256, 0, stream>>>(v2, inv2);

    if (ws_size >= needF16) {
        // f16x3 split path (primary): ~2.7x fewer WMMA issues at full-rate f16
        _Float16* a_hi = (_Float16*)(w + smallFloats);
        _Float16* a_lo = a_hi + halfElems;
        _Float16* b_hi = a_lo + halfElems;
        _Float16* b_lo = b_hi + halfElems;
        const int nb4 = (int)(halfElems / 4 / 256);      // one float4 per thread
        k_split<<<nb4, 256, 0, stream>>>(v1, a_hi, a_lo);
        k_split<<<nb4, 256, 0, stream>>>(v2, b_hi, b_lo);
        k_gemm_lse_f16<<<n / BM, 256, 0, stream>>>(a_hi, a_lo, b_hi, b_lo,
                                                   inv1, inv2, lse, n);
    } else {
        // fp32 WMMA fallback when scratch is tight
        k_gemm_lse_f32<<<n / BM, 256, 0, stream>>>(v1, v2, inv1, inv2, lse, n);
    }

    k_diag_loss<<<n / 8, 256, 0, stream>>>(v1, v2, inv1, inv2, lse, partial);
    k_final<<<1, 256, 0, stream>>>(partial, (float*)d_out, n / 8, n);
}
